// GAT_27255862460801
// MI455X (gfx1250) — compile-verified, matrix operand order checked by reading, not run on hardware
//
#include <hip/hip_runtime.h>
#include <hip/hip_bf16.h>
#include <stdint.h>

// ---------------------------------------------------------------------------
// GAT (3-layer) for MI455X / gfx1250.
// Dense projections: v_wmma_f32_16x16x32_bf16, one wave per 16x128 row-block
// (8 accumulators, A fragment reused across 8 column tiles, 32 WMMAs/wave).
// Edge phase: wave-per-edge float4 gather + global_atomic_add_f32 scatter.
// ---------------------------------------------------------------------------

#define NNODES 50000
#define NEDGES 800000
#define NFEAT  128        // IN == H*HID == 128 throughout
#define HEADS0 4
#define CH0    32
#define HEADS2 6
#define CH2    2
#define NEG_SLOPE 0.2f

typedef __attribute__((ext_vector_type(16))) __bf16 v16bf;
typedef __attribute__((ext_vector_type(8)))  float  v8f;

union FragAB { v16bf v; unsigned u[8]; };

__device__ __forceinline__ unsigned short f2bf(float f) {
    unsigned u = __float_as_uint(f);
    unsigned r = u + 0x7FFFu + ((u >> 16) & 1u);   // round-to-nearest-even
    return (unsigned short)(r >> 16);
}

// --------------------------- conversion kernels ----------------------------

__global__ void k_f32_to_bf16(const float* __restrict__ in,
                              unsigned short* __restrict__ out, int n) {
    int i = blockIdx.x * blockDim.x + threadIdx.x;
    if (i < n) out[i] = f2bf(in[i]);
}

// W is [128, ncols] row-major f32. Produce Wt[n*128 + k] bf16, zero-padded to ncp rows.
__global__ void k_convert_W_T(const float* __restrict__ W,
                              unsigned short* __restrict__ Wt,
                              int ncols, int ncp) {
    int i = blockIdx.x * blockDim.x + threadIdx.x;
    if (i >= ncp * NFEAT) return;
    int n = i / NFEAT;
    int k = i % NFEAT;
    Wt[i] = (n < ncols) ? f2bf(W[k * ncols + n]) : (unsigned short)0;
}

__global__ void k_fill(float* __restrict__ p, float v, int n) {
    int i = blockIdx.x * blockDim.x + threadIdx.x;
    if (i < n) p[i] = v;
}

// ------------------------------ WMMA GEMM ----------------------------------
// C[M, ncols] = A(bf16 [M,128]) x W(via Wt bf16 [16*CT,128])
// One wave computes a 16 x (16*CT) output block: the A fragment for each
// K-step is loaded once and reused across CT column tiles (CT WMMAs/K-step).
template <int CT>
__global__ __launch_bounds__(256)
void k_gemm_wmma(const unsigned short* __restrict__ Abf,
                 const unsigned short* __restrict__ Bt,
                 float* __restrict__ C,
                 int ncols, int ldc, int rowTiles) {
    int wave = threadIdx.x >> 5;
    int lane = threadIdx.x & 31;
    int rt = blockIdx.x * (blockDim.x >> 5) + wave;
    if (rt >= rowTiles) return;                  // wave-uniform: EXEC all-ones
    int rowBase = rt << 4;
    int kb = (lane < 16) ? 0 : 8;                // K base per ISA 16-bit A layout
    int ln = lane & 15;

    const unsigned* __restrict__ Ap = (const unsigned*)(Abf + (size_t)(rowBase + ln) * NFEAT);

    v8f acc[CT];
#pragma unroll
    for (int ct = 0; ct < CT; ct++) acc[ct] = (v8f){};

#pragma unroll
    for (int k0 = 0; k0 < NFEAT; k0 += 32) {
        FragAB a;
#pragma unroll
        for (int i = 0; i < 8; i++) {
            int k = k0 + ((i & 3) << 1) + ((i & 4) << 2) + kb;  // VGPR i -> packed K pair
            a.u[i] = Ap[k >> 1];
        }
#pragma unroll
        for (int ct = 0; ct < CT; ct++) {
            const unsigned* __restrict__ Bp =
                (const unsigned*)(Bt + (size_t)(ct * 16 + ln) * NFEAT);
            FragAB b;
#pragma unroll
            for (int i = 0; i < 8; i++) {
                int k = k0 + ((i & 3) << 1) + ((i & 4) << 2) + kb;
                b.u[i] = Bp[k >> 1];
            }
            acc[ct] = __builtin_amdgcn_wmma_f32_16x16x32_bf16(
                false, a.v, false, b.v, (short)0, acc[ct], false, false);
        }
    }

    int mb = (lane < 16) ? 0 : 8;                 // C/D layout: VGPR r -> M = r + mb
#pragma unroll
    for (int ct = 0; ct < CT; ct++) {
        int col = ct * 16 + ln;
        if (col < ncols) {
#pragma unroll
            for (int r = 0; r < 8; r++)
                C[(size_t)(rowBase + mb + r) * ldc + col] = acc[ct][r];
        }
    }
}

// --------------------------- attention pipeline ----------------------------

// al_s[n,h] = sum_c xw[n, h*ch + c] * a_s[h*ch + c]   (likewise al_d)
__global__ void k_attn_logits(const float* __restrict__ xw,
                              const float* __restrict__ a_s,
                              const float* __restrict__ a_d,
                              float* __restrict__ al_s,
                              float* __restrict__ al_d,
                              int heads, int ch) {
    int i = blockIdx.x * blockDim.x + threadIdx.x;
    if (i >= NNODES * heads) return;
    int n = i / heads, h = i % heads;
    const float* row = xw + (size_t)n * heads * ch + h * ch;
    float ss = 0.f, sd = 0.f;
    for (int c = 0; c < ch; c++) {
        float v = row[c];
        ss += v * a_s[h * ch + c];
        sd += v * a_d[h * ch + c];
    }
    al_s[i] = ss;
    al_d[i] = sd;
}

// e = exp(leaky_relu(al_s[src] + al_d[dst])); segment-sum into s_sum[dst].
// Softmax is shift-invariant; logits are O(1) here so max-subtraction is skipped.
__global__ void k_edge_exp(const int* __restrict__ ei,
                           const float* __restrict__ al_s,
                           const float* __restrict__ al_d,
                           float* __restrict__ e_edge,
                           float* __restrict__ s_sum,
                           int heads) {
    int i = blockIdx.x * blockDim.x + threadIdx.x;
    int tot = (NEDGES + NNODES) * heads;
    if (i >= tot) return;
    int e = i / heads, h = i % heads;
    int src, dst;
    if (e < NEDGES) { src = ei[e]; dst = ei[NEDGES + e]; }
    else            { src = dst = e - NEDGES; }            // self-loops
    float a = al_s[src * heads + h] + al_d[dst * heads + h];
    a = (a > 0.f) ? a : NEG_SLOPE * a;
    float ex = __expf(a);
    e_edge[i] = ex;
    atomicAdd(&s_sum[dst * heads + h], ex);
}

// Concat layers (C=128, heads=4): one wave per edge. Edge indices are
// wave-uniform (scalar loads); lane l owns channels 4l..4l+3 -> head = l/8,
// one float4 gather + 4 atomic f32 scatters per lane.
__global__ __launch_bounds__(256)
void k_aggregate128(const int* __restrict__ ei,
                    const float* __restrict__ xw,
                    const float* __restrict__ e_edge,
                    const float* __restrict__ s_sum,
                    float* __restrict__ acc) {
    int wave = threadIdx.x >> 5;
    int lane = threadIdx.x & 31;
    int e = blockIdx.x * (blockDim.x >> 5) + wave;
    if (e >= NEDGES + NNODES) return;
    int src, dst;
    if (e < NEDGES) { src = ei[e]; dst = ei[NEDGES + e]; }
    else            { src = dst = e - NEDGES; }
    int h = lane >> 3;                                   // (4*lane)/32
    float coef = e_edge[(size_t)e * HEADS0 + h] / s_sum[dst * HEADS0 + h];
    const float4 v = *(const float4*)(xw + (size_t)src * NFEAT + 4 * lane);
    float* d = acc + (size_t)dst * NFEAT + 4 * lane;
    atomicAdd(d + 0, v.x * coef);
    atomicAdd(d + 1, v.y * coef);
    atomicAdd(d + 2, v.z * coef);
    atomicAdd(d + 3, v.w * coef);
}

// Generic scalar aggregation (used for the small final layer, C=12).
__global__ void k_aggregate(const int* __restrict__ ei,
                            const float* __restrict__ xw,
                            const float* __restrict__ e_edge,
                            const float* __restrict__ s_sum,
                            float* __restrict__ acc,
                            int heads, int ch) {
    int i = blockIdx.x * blockDim.x + threadIdx.x;
    int C = heads * ch;
    int tot = (NEDGES + NNODES) * C;
    if (i >= tot) return;
    int e = i / C, c = i % C, h = c / ch;
    int src, dst;
    if (e < NEDGES) { src = ei[e]; dst = ei[NEDGES + e]; }
    else            { src = dst = e - NEDGES; }
    float coef = e_edge[(size_t)e * heads + h] / s_sum[dst * heads + h];
    atomicAdd(&acc[(size_t)dst * C + c], xw[(size_t)src * C + c] * coef);
}

// acc holds the Linear result; add l_b[c] + c_b[c] before aggregation adds in.
__global__ void k_add_bias(float* __restrict__ acc,
                           const float* __restrict__ lb,
                           const float* __restrict__ cb, int C) {
    int i = blockIdx.x * blockDim.x + threadIdx.x;
    if (i >= NNODES * C) return;
    int c = i % C;
    acc[i] += lb[c] + cb[c];
}

__global__ void k_relu(const float* __restrict__ in, float* __restrict__ out, int n) {
    int i = blockIdx.x * blockDim.x + threadIdx.x;
    if (i < n) out[i] = fmaxf(in[i], 0.f);
}

// Final layer: mean over 6 heads of acc2[N,12] + c2_b + lin2[N,2] + l2_b
__global__ void k_final(const float* __restrict__ acc2,
                        const float* __restrict__ lin2,
                        const float* __restrict__ c2b,
                        const float* __restrict__ l2b,
                        float* __restrict__ out) {
    int i = blockIdx.x * blockDim.x + threadIdx.x;
    if (i >= NNODES * 2) return;
    int n = i >> 1, o = i & 1;
    float s = 0.f;
#pragma unroll
    for (int h = 0; h < HEADS2; h++) s += acc2[n * 12 + h * 2 + o];
    out[i] = s * (1.f / 6.f) + c2b[o] + lin2[i] + l2b[o];
}

// ------------------------------- launcher ----------------------------------

static inline int cdiv(int a, int b) { return (a + b - 1) / b; }

extern "C" void kernel_launch(void* const* d_in, const int* in_sizes, int n_in,
                              void* d_out, int out_size, void* d_ws, size_t ws_size,
                              hipStream_t stream) {
    const float* x     = (const float*)d_in[0];
    const int*   ei    = (const int*)d_in[1];
    // d_in[2] = edge_attr : ignored (GATConv edge_dim=None)
    const float* c0_W  = (const float*)d_in[3];
    const float* c0_as = (const float*)d_in[4];
    const float* c0_ad = (const float*)d_in[5];
    const float* c0_b  = (const float*)d_in[6];
    const float* l0_W  = (const float*)d_in[7];
    const float* l0_b  = (const float*)d_in[8];
    const float* c1_W  = (const float*)d_in[9];
    const float* c1_as = (const float*)d_in[10];
    const float* c1_ad = (const float*)d_in[11];
    const float* c1_b  = (const float*)d_in[12];
    const float* l1_W  = (const float*)d_in[13];
    const float* l1_b  = (const float*)d_in[14];
    const float* c2_W  = (const float*)d_in[15];
    const float* c2_as = (const float*)d_in[16];
    const float* c2_ad = (const float*)d_in[17];
    const float* c2_b  = (const float*)d_in[18];
    const float* l2_W  = (const float*)d_in[19];
    const float* l2_b  = (const float*)d_in[20];
    float* out = (float*)d_out;

    // --- carve workspace ---
    size_t off = 0;
    auto carve = [&](size_t bytes) -> void* {
        void* p = (char*)d_ws + off;
        off += (bytes + 255) & ~(size_t)255;
        return p;
    };
    const int ETOT = NEDGES + NNODES;
    float*          hf    = (float*)         carve((size_t)NNODES * NFEAT * 4);
    unsigned short* hbf   = (unsigned short*)carve((size_t)NNODES * NFEAT * 2);
    float*          xw    = (float*)         carve((size_t)NNODES * NFEAT * 4);
    float*          acc   = (float*)         carve((size_t)NNODES * NFEAT * 4);
    unsigned short* Wt    = (unsigned short*)carve((size_t)NFEAT * NFEAT * 2);
    float*          al_s  = (float*)         carve((size_t)NNODES * 6 * 4);
    float*          al_d  = (float*)         carve((size_t)NNODES * 6 * 4);
    float*          s_sum = (float*)         carve((size_t)NNODES * 6 * 4);
    float*          e_edg = (float*)         carve((size_t)ETOT * 6 * 4);
    float*          acc2  = (float*)         carve((size_t)NNODES * 12 * 4);
    float*          lin2  = (float*)         carve((size_t)NNODES * 2 * 4);

    const int B = 256;
    const int rowTiles = NNODES / 16;               // 3125, exact
    const int gemmBlocks = cdiv(rowTiles, 8);       // 8 waves/block
    const int nElem = NNODES * NFEAT;

    // --------- layers 0 and 1 (identical shapes) ---------
    auto concat_layer = [&](const float* featF32,
                            const float* cW, const float* as, const float* ad,
                            const float* cb, const float* lW, const float* lb) {
        k_f32_to_bf16<<<cdiv(nElem, B), B, 0, stream>>>(featF32, hbf, nElem);
        k_convert_W_T<<<cdiv(NFEAT * NFEAT, B), B, 0, stream>>>(cW, Wt, NFEAT, NFEAT);
        k_gemm_wmma<8><<<gemmBlocks, B, 0, stream>>>(hbf, Wt, xw, NFEAT, NFEAT, rowTiles);
        k_convert_W_T<<<cdiv(NFEAT * NFEAT, B), B, 0, stream>>>(lW, Wt, NFEAT, NFEAT);
        k_gemm_wmma<8><<<gemmBlocks, B, 0, stream>>>(hbf, Wt, acc, NFEAT, NFEAT, rowTiles);
        k_add_bias<<<cdiv(nElem, B), B, 0, stream>>>(acc, lb, cb, NFEAT);
        k_attn_logits<<<cdiv(NNODES * HEADS0, B), B, 0, stream>>>(xw, as, ad, al_s, al_d, HEADS0, CH0);
        k_fill<<<cdiv(NNODES * HEADS0, B), B, 0, stream>>>(s_sum, 0.f, NNODES * HEADS0);
        k_edge_exp<<<cdiv(ETOT * HEADS0, B), B, 0, stream>>>(ei, al_s, al_d, e_edg, s_sum, HEADS0);
        k_aggregate128<<<cdiv(ETOT, 8), B, 0, stream>>>(ei, xw, e_edg, s_sum, acc);
        k_relu<<<cdiv(nElem, B), B, 0, stream>>>(acc, hf, nElem);
    };

    concat_layer(x,  c0_W, c0_as, c0_ad, c0_b, l0_W, l0_b);
    concat_layer(hf, c1_W, c1_as, c1_ad, c1_b, l1_W, l1_b);

    // --------- layer 2 (heads=6, ch=2, concat=False -> mean) ---------
    k_f32_to_bf16<<<cdiv(nElem, B), B, 0, stream>>>(hf, hbf, nElem);
    k_convert_W_T<<<cdiv(16 * NFEAT, B), B, 0, stream>>>(c2_W, Wt, 12, 16);
    k_gemm_wmma<1><<<gemmBlocks, B, 0, stream>>>(hbf, Wt, xw, 12, 12, rowTiles);
    k_convert_W_T<<<cdiv(16 * NFEAT, B), B, 0, stream>>>(l2_W, Wt, 2, 16);
    k_gemm_wmma<1><<<gemmBlocks, B, 0, stream>>>(hbf, Wt, lin2, 2, 2, rowTiles);
    k_attn_logits<<<cdiv(NNODES * HEADS2, B), B, 0, stream>>>(xw, c2_as, c2_ad, al_s, al_d, HEADS2, CH2);
    k_fill<<<cdiv(NNODES * HEADS2, B), B, 0, stream>>>(s_sum, 0.f, NNODES * HEADS2);
    k_fill<<<cdiv(NNODES * 12, B), B, 0, stream>>>(acc2, 0.f, NNODES * 12);
    k_edge_exp<<<cdiv(ETOT * HEADS2, B), B, 0, stream>>>(ei, al_s, al_d, e_edg, s_sum, HEADS2);
    k_aggregate<<<cdiv(ETOT * 12, B), B, 0, stream>>>(ei, xw, e_edg, s_sum, acc2, HEADS2, CH2);
    k_final<<<cdiv(NNODES * 2, B), B, 0, stream>>>(acc2, lin2, c2_b, l2_b, out);

    (void)in_sizes; (void)n_in; (void)out_size; (void)ws_size;
}
